// ModifiedGATLayer_12781822673343
// MI455X (gfx1250) — matrix operand order repaired and checked by reading, not compiled, not run on hardware
//
#include <hip/hip_runtime.h>
#include <hip/hip_bf16.h>
#include <math.h>

// ---------------------------------------------------------------------------
// Problem constants
// ---------------------------------------------------------------------------
#define NROWS 8192
#define DF    512
#define QLD   520          // padded LDS row stride (elems) for Rs/Qs: bank step 4
#define PLD   72           // padded LDS row stride (elems) for Pbuf: bank step 36

typedef __bf16 bf16_t;
typedef __attribute__((ext_vector_type(16))) __bf16 bf16x16;
typedef __attribute__((ext_vector_type(8)))  float  f32x8;

// ---------------------------------------------------------------------------
// Helpers
// ---------------------------------------------------------------------------
static __device__ __forceinline__ bf16_t f2bf(float f) {
    unsigned u = __builtin_bit_cast(unsigned, f);
    unsigned r = (u + 0x7FFFu + ((u >> 16) & 1u)) >> 16;   // round-nearest-even
    return __builtin_bit_cast(bf16_t, (unsigned short)r);
}

// A fragment: 16x32 bf16 (MxK) from row-major matrix (stride ld elements).
static __device__ __forceinline__ bf16x16
load_frag_a(const bf16_t* base, int ld, int row0, int k0, int lane) {
    int r = lane & 15, h = lane >> 4;
    const bf16_t* p = base + (size_t)(row0 + r) * ld + k0 + h * 8;
    bf16x16 f;
    *reinterpret_cast<uint4*>(&f)       = *reinterpret_cast<const uint4*>(p);
    *(reinterpret_cast<uint4*>(&f) + 1) = *reinterpret_cast<const uint4*>(p + 16);
    return f;
}

// B fragment: 32x16 bf16 (KxN); source rows are B's columns (stride ld).
static __device__ __forceinline__ bf16x16
load_frag_b(const bf16_t* base, int ld, int col0, int k0, int lane) {
    int r = lane & 15, h = lane >> 4;
    const bf16_t* p = base + (size_t)(col0 + r) * ld + k0 + h * 8;
    bf16x16 f;
    *reinterpret_cast<uint4*>(&f)       = *reinterpret_cast<const uint4*>(p);
    *(reinterpret_cast<uint4*>(&f) + 1) = *reinterpret_cast<const uint4*>(p + 16);
    return f;
}

static __device__ __forceinline__ f32x8 wmma_bf16(bf16x16 a, bf16x16 b, f32x8 c) {
    return __builtin_amdgcn_wmma_f32_16x16x32_bf16(false, a, false, b,
                                                   (short)0, c, false, false);
}

static __device__ __forceinline__ float redmax16(float v) {
    v = fmaxf(v, __shfl_xor(v, 1, 32));
    v = fmaxf(v, __shfl_xor(v, 2, 32));
    v = fmaxf(v, __shfl_xor(v, 4, 32));
    v = fmaxf(v, __shfl_xor(v, 8, 32));
    return v;
}
static __device__ __forceinline__ float redsum16(float v) {
    v += __shfl_xor(v, 1, 32);
    v += __shfl_xor(v, 2, 32);
    v += __shfl_xor(v, 4, 32);
    v += __shfl_xor(v, 8, 32);
    return v;
}

// Async copy of a 64x512 bf16 tile (row-major, global) into LDS with padded
// stride QLD. 256 threads x 16 x b128 lanes-worth = 64 KB; tracked by ASYNCcnt.
static __device__ __forceinline__ void
async_copy_tile(const bf16_t* __restrict__ gsrc, bf16_t* lds_dst, int tid) {
#pragma unroll
    for (int i = 0; i < 16; ++i) {
        int c = i * 256 + tid;              // 4096 chunks of 8 elems
        int row = c >> 6, col8 = c & 63;
        const bf16_t* gp = gsrc + (size_t)row * DF + col8 * 8;
        unsigned lo = (unsigned)(size_t)(lds_dst + row * QLD + col8 * 8);
        asm volatile("global_load_async_to_lds_b128 %0, %1, off"
                     :: "v"(lo), "v"(gp) : "memory");
    }
}
static __device__ __forceinline__ void wait_async0() {
    asm volatile("s_wait_asynccnt 0x0" ::: "memory");
}

// ---------------------------------------------------------------------------
// Small prep kernels (run once per launch; ~0.4 GFLOP total)
// ---------------------------------------------------------------------------
__global__ void cvt_bf16_kernel(const float* __restrict__ src,
                                bf16_t* __restrict__ dst, int n) {
    int i = blockIdx.x * 256 + threadIdx.x;
    if (i < n) dst[i] = f2bf(src[i]);
}

// Weff[o][b] = sum_a Wl1[o][a]*W3[a][b][1] + sum_a Wl2[o][a]*W5[a][b][2] + Wl3[o][b]
__global__ void weff_kernel(const float* __restrict__ Wl,
                            const float* __restrict__ W3,
                            const float* __restrict__ W5,
                            float* __restrict__ Weff) {
    int idx = blockIdx.x * 256 + threadIdx.x;     // 512*512
    int o = idx >> 9, b = idx & 511;
    float s = Wl[o * 1536 + 1024 + b];
    for (int a = 0; a < 512; ++a) {
        s += Wl[o * 1536 + a]       * W3[a * 1536 + b * 3 + 1];
        s += Wl[o * 1536 + 512 + a] * W5[a * 2560 + b * 5 + 2];
    }
    Weff[idx] = s;
}

__global__ void beff_kernel(const float* __restrict__ Wl,
                            const float* __restrict__ b3,
                            const float* __restrict__ b5,
                            const float* __restrict__ bl,
                            float* __restrict__ beff) {
    int o = blockIdx.x * 256 + threadIdx.x;
    if (o >= 512) return;
    float s = bl[o];
    for (int a = 0; a < 512; ++a) {
        s += Wl[o * 1536 + a] * b3[a] + Wl[o * 1536 + 512 + a] * b5[a];
    }
    beff[o] = s;
}

// Wkeff[o][i] = sum_b Weff[o][b] * Wk[b][i]   (written directly as bf16)
__global__ void wkeff_kernel(const float* __restrict__ Weff,
                             const float* __restrict__ Wk,
                             bf16_t* __restrict__ Wk2b) {
    int idx = blockIdx.x * 256 + threadIdx.x;
    int o = idx >> 9, i = idx & 511;
    float s = 0.f;
    for (int b = 0; b < 512; ++b) s += Weff[o * 512 + b] * Wk[b * 512 + i];
    Wk2b[idx] = f2bf(s);
}

__global__ void bkeff_kernel(const float* __restrict__ Weff,
                             const float* __restrict__ bk,
                             const float* __restrict__ beff,
                             float* __restrict__ bkeff) {
    int o = blockIdx.x * 256 + threadIdx.x;
    if (o >= 512) return;
    float s = beff[o];
    for (int b = 0; b < 512; ++b) s += Weff[o * 512 + b] * bk[b];
    bkeff[o] = s;
}

// ---------------------------------------------------------------------------
// Fused projection GEMM: out = x @ W.T + b  (bf16 WMMA, fp32 accum)
// blockIdx.z selects {Q, K_new, V}; V is stored transposed as Vt[512][8192].
// ---------------------------------------------------------------------------
__global__ __launch_bounds__(256, 1)
void qkv_gemm_kernel(const bf16_t* __restrict__ xb,
                     const bf16_t* __restrict__ Wqb,
                     const bf16_t* __restrict__ Wk2b,
                     const bf16_t* __restrict__ Wvb,
                     const float* __restrict__ bq,
                     const float* __restrict__ bk2,
                     const float* __restrict__ bv,
                     bf16_t* __restrict__ Qb,
                     bf16_t* __restrict__ Rb,
                     bf16_t* __restrict__ Vtb) {
    int tid = threadIdx.x, lane = tid & 31, w = tid >> 5;
    int r = lane & 15, h = lane >> 4;
    int sel = blockIdx.z;
    const bf16_t* W    = (sel == 0) ? Wqb : (sel == 1) ? Wk2b : Wvb;
    const float*  bias = (sel == 0) ? bq  : (sel == 1) ? bk2  : bv;

    int m0 = blockIdx.x * 64;
    int n0 = blockIdx.y * 64;
    int mi  = w >> 1;                 // row tile 0..3
    int nt0 = (w & 1) * 2;            // first col tile

    f32x8 c0 = {0,0,0,0,0,0,0,0}, c1 = c0;
    for (int k = 0; k < DF; k += 32) {
        bf16x16 a  = load_frag_a(xb, DF, m0 + mi * 16, k, lane);
        bf16x16 b0 = load_frag_b(W,  DF, n0 + nt0 * 16,        k, lane);
        bf16x16 b1 = load_frag_b(W,  DF, n0 + (nt0 + 1) * 16,  k, lane);
        c0 = wmma_bf16(a, b0, c0);
        c1 = wmma_bf16(a, b1, c1);
    }
#pragma unroll
    for (int t = 0; t < 2; ++t) {
        f32x8 c = t ? c1 : c0;
        int col = n0 + (nt0 + t) * 16 + r;
        float bb = bias[col];
#pragma unroll
        for (int j = 0; j < 8; ++j) {
            int row = m0 + mi * 16 + h * 8 + j;
            bf16_t v = f2bf(c[j] + bb);
            if (sel == 0)      Qb[(size_t)row * DF + col] = v;
            else if (sel == 1) Rb[(size_t)row * DF + col] = v;
            else               Vtb[(size_t)col * NROWS + row] = v;  // transposed
        }
    }
}

// ---------------------------------------------------------------------------
// Flash attention: out[b] = softmax_i(K_new[b].Q[i]/sqrt(D)) @ V
// "queries" = K_new rows (Rb), "keys" = Q rows (Qb), values = Vt.
// R block staged once in LDS; Q key-tiles double-buffered in LDS via
// GLOBAL_LOAD_ASYNC_TO_LDS (ASYNCcnt) with prefetch overlapping the WMMAs.
// ---------------------------------------------------------------------------
__global__ __launch_bounds__(256, 1)
void attn_kernel(const bf16_t* __restrict__ Rb,
                 const bf16_t* __restrict__ Qb,
                 const bf16_t* __restrict__ Vtb,
                 float* __restrict__ out) {
    __shared__ float m_state[64], l_state[64];
    __shared__ float pmax[64][2], psum[64][2];
    __shared__ float mnew_s[64], rscale_s[64];
    __shared__ __align__(16) bf16_t Rs[64 * QLD];        // 65 KB, loop-invariant
    __shared__ __align__(16) bf16_t Qs[2][64 * QLD];     // 130 KB, double buffer
    __shared__ __align__(16) bf16_t Pbuf[64 * PLD];      // 9 KB

    int tid = threadIdx.x, lane = tid & 31, w = tid >> 5;
    int r = lane & 15, h = lane >> 4;
    int rowBase = blockIdx.x * 64;

    int smi   = w >> 1;       // S-phase row tile (0..3)
    int spart = w & 1;        // S-phase key half (0/1)
    const float sscale = 0.04419417382415922f;   // 1/sqrt(512)

    if (tid < 64) { m_state[tid] = -3.0e38f; l_state[tid] = 0.f; }

    // Stage loop-invariant R block + first Q tile (async, ASYNCcnt-tracked).
    async_copy_tile(Rb + (size_t)rowBase * DF, Rs, tid);
    async_copy_tile(Qb, Qs[0], tid);

    f32x8 acc[4][4];
#pragma unroll
    for (int i = 0; i < 4; ++i)
#pragma unroll
        for (int j = 0; j < 4; ++j) acc[i][j] = (f32x8){0,0,0,0,0,0,0,0};

    for (int it = 0; it < NROWS / 64; ++it) {
        int kt = it * 64;
        int cur = it & 1;

        // Current tile resident; all waves agree; prefetch next tile.
        wait_async0();
        __syncthreads();
        if (it + 1 < NROWS / 64)
            async_copy_tile(Qb + (size_t)(kt + 64) * DF, Qs[cur ^ 1], tid);

        // ---- S = R_block @ Q_tile^T (LDS operands) ---------------------
        f32x8 s0 = {0,0,0,0,0,0,0,0}, s1 = s0;
        const bf16_t* qs = Qs[cur];
        for (int k = 0; k < DF; k += 32) {
            bf16x16 a  = load_frag_a(Rs, QLD, smi * 16, k, lane);
            bf16x16 b0 = load_frag_b(qs, QLD, spart * 32,      k, lane);
            bf16x16 b1 = load_frag_b(qs, QLD, spart * 32 + 16, k, lane);
            s0 = wmma_bf16(a, b0, s0);
            s1 = wmma_bf16(a, b1, s1);
        }
        float mv[8];
#pragma unroll
        for (int j = 0; j < 8; ++j) {
            s0[j] *= sscale; s1[j] *= sscale;
            mv[j] = redmax16(fmaxf(s0[j], s1[j]));
        }
        if (r == 0) {
#pragma unroll
            for (int j = 0; j < 8; ++j) pmax[smi * 16 + h * 8 + j][spart] = mv[j];
        }
        __syncthreads();

        // ---- online-softmax row stats ----------------------------------
        if (tid < 64) {
            float mo = m_state[tid];
            float mn = fmaxf(mo, fmaxf(pmax[tid][0], pmax[tid][1]));
            mnew_s[tid]  = mn;
            rscale_s[tid] = __expf(mo - mn);
            m_state[tid] = mn;
        }
        __syncthreads();

        // ---- P = exp(S - m), row sums, stage P (bf16) in LDS -----------
        float ps[8];
#pragma unroll
        for (int j = 0; j < 8; ++j) {
            int row = smi * 16 + h * 8 + j;
            float mrow = mnew_s[row];
            float p0 = __expf(s0[j] - mrow);
            float p1 = __expf(s1[j] - mrow);
            Pbuf[row * PLD + spart * 32 + r]      = f2bf(p0);
            Pbuf[row * PLD + spart * 32 + 16 + r] = f2bf(p1);
            ps[j] = redsum16(p0 + p1);
        }
        if (r == 0) {
#pragma unroll
            for (int j = 0; j < 8; ++j) psum[smi * 16 + h * 8 + j][spart] = ps[j];
        }
        // rescale running output accumulator
#pragma unroll
        for (int mi2 = 0; mi2 < 4; ++mi2)
#pragma unroll
            for (int j = 0; j < 8; ++j) {
                float sc = rscale_s[mi2 * 16 + h * 8 + j];
#pragma unroll
                for (int di = 0; di < 4; ++di) acc[mi2][di][j] *= sc;
            }
        __syncthreads();

        if (tid < 64)
            l_state[tid] = l_state[tid] * rscale_s[tid] + psum[tid][0] + psum[tid][1];

        // ---- O += P @ V_tile (wave w owns D-cols [64w, 64w+64)) --------
#pragma unroll
        for (int kk = 0; kk < 64; kk += 32) {
            bf16x16 bfr[4];
#pragma unroll
            for (int di = 0; di < 4; ++di)
                bfr[di] = load_frag_b(Vtb, NROWS, w * 64 + di * 16, kt + kk, lane);
#pragma unroll
            for (int mi2 = 0; mi2 < 4; ++mi2) {
                bf16x16 afr = load_frag_a(Pbuf, PLD, mi2 * 16, kk, lane);
#pragma unroll
                for (int di = 0; di < 4; ++di)
                    acc[mi2][di] = wmma_bf16(afr, bfr[di], acc[mi2][di]);
            }
        }
        __syncthreads();
    }

    // ---- normalize by 1/l and store fp32 -------------------------------
#pragma unroll
    for (int mi2 = 0; mi2 < 4; ++mi2)
#pragma unroll
        for (int j = 0; j < 8; ++j) {
            float inv = 1.0f / l_state[mi2 * 16 + h * 8 + j];
            int row = rowBase + mi2 * 16 + h * 8 + j;
#pragma unroll
            for (int di = 0; di < 4; ++di)
                out[(size_t)row * DF + w * 64 + di * 16 + r] = acc[mi2][di][j] * inv;
        }
}

// ---------------------------------------------------------------------------
// Host launcher
// ---------------------------------------------------------------------------
extern "C" void kernel_launch(void* const* d_in, const int* in_sizes, int n_in,
                              void* d_out, int out_size, void* d_ws, size_t ws_size,
                              hipStream_t stream) {
    const float* x  = (const float*)d_in[0];
    const float* Wq = (const float*)d_in[1];
    const float* bq = (const float*)d_in[2];
    const float* Wk = (const float*)d_in[3];
    const float* bk = (const float*)d_in[4];
    const float* Wv = (const float*)d_in[5];
    const float* bv = (const float*)d_in[6];
    const float* W3 = (const float*)d_in[7];
    const float* b3 = (const float*)d_in[8];
    const float* W5 = (const float*)d_in[9];
    const float* b5 = (const float*)d_in[10];
    const float* Wl = (const float*)d_in[11];
    const float* bl = (const float*)d_in[12];
    float* out = (float*)d_out;

    char* ws = (char*)d_ws;
    size_t off = 0;
    bf16_t* xb    = (bf16_t*)(ws + off); off += (size_t)NROWS * DF * 2;   // 8 MB
    bf16_t* Qb    = (bf16_t*)(ws + off); off += (size_t)NROWS * DF * 2;   // 8 MB
    bf16_t* Rb    = (bf16_t*)(ws + off); off += (size_t)NROWS * DF * 2;   // 8 MB
    bf16_t* Vtb   = (bf16_t*)(ws + off); off += (size_t)NROWS * DF * 2;   // 8 MB
    bf16_t* Wqb   = (bf16_t*)(ws + off); off += (size_t)DF * DF * 2;
    bf16_t* Wvb   = (bf16_t*)(ws + off); off += (size_t)DF * DF * 2;
    bf16_t* Wk2b  = (bf16_t*)(ws + off); off += (size_t)DF * DF * 2;
    float*  Weff  = (float*)(ws + off);  off += (size_t)DF * DF * 4;
    float*  beff  = (float*)(ws + off);  off += 2048;
    float*  bkeff = (float*)(ws + off);  off += 2048;

    // 1) convert activations & static weights to bf16
    cvt_bf16_kernel<<<(NROWS * DF + 255) / 256, 256, 0, stream>>>(x, xb, NROWS * DF);
    cvt_bf16_kernel<<<(DF * DF + 255) / 256, 256, 0, stream>>>(Wq, Wqb, DF * DF);
    cvt_bf16_kernel<<<(DF * DF + 255) / 256, 256, 0, stream>>>(Wv, Wvb, DF * DF);

    // 2) fold conv-taps + concat-linear + K-projection into one 512x512 weight
    weff_kernel<<<(DF * DF) / 256, 256, 0, stream>>>(Wl, W3, W5, Weff);
    beff_kernel<<<2, 256, 0, stream>>>(Wl, b3, b5, bl, beff);
    wkeff_kernel<<<(DF * DF) / 256, 256, 0, stream>>>(Weff, Wk, Wk2b);
    bkeff_kernel<<<2, 256, 0, stream>>>(Weff, bk, beff, bkeff);

    // 3) Q / K_new / V projections via bf16 WMMA (V stored transposed)
    qkv_gemm_kernel<<<dim3(NROWS / 64, DF / 64, 3), 256, 0, stream>>>(
        xb, Wqb, Wk2b, Wvb, bq, bkeff, bv, Qb, Rb, Vtb);

    // 4) fused flash attention (async-LDS staged, double-buffered)
    attn_kernel<<<NROWS / 64, 256, 0, stream>>>(Rb, Qb, Vtb, out);
}
